// T5SelfAttention_27384711480094
// MI455X (gfx1250) — compile-verified, matrix operand order checked by reading, not run on hardware
//
#include <hip/hip_runtime.h>

// ---------------------------------------------------------------------------
// Types
// ---------------------------------------------------------------------------
typedef __attribute__((ext_vector_type(16))) __bf16 v16bf;
typedef __attribute__((ext_vector_type(8)))  float  v8f;
typedef __attribute__((ext_vector_type(4)))  float  v4f;
typedef __attribute__((ext_vector_type(4)))  unsigned int v4u;

union RB { v4u u; __bf16 e[8]; };

#define SEQ    2048
#define DHEAD  64
#define NHEAD  16
#define DMODEL 1024
#define NROWS  4096            // 2 batches * 2048 seq

// ---------------------------------------------------------------------------
// CDNA5 async global->LDS helpers (ASYNCcnt-tracked DMA, no VGPR round trip)
// ---------------------------------------------------------------------------
__device__ __forceinline__ unsigned lds_off(const void* p) {
  // generic pointer to LDS: low 32 bits are the LDS byte offset (ISA 10.2)
  return (unsigned)(unsigned long long)p;
}
__device__ __forceinline__ void async_ld_b128(unsigned lds, const void* gaddr) {
  asm volatile("global_load_async_to_lds_b128 %0, %1, off"
               :: "v"(lds), "v"(gaddr) : "memory");
}
__device__ __forceinline__ void wait_async0() {
  asm volatile("s_wait_asynccnt 0" ::: "memory");
}

// ---------------------------------------------------------------------------
// WMMA helpers
// ---------------------------------------------------------------------------
__device__ __forceinline__ v8f wmma_bf16(v16bf a, v16bf b, v8f c) {
  return __builtin_amdgcn_wmma_f32_16x16x32_bf16(false, a, false, b, (short)0, c,
                                                 false, false);
}

// 16x32 A-fragment (or B-fragment from transposed operand) from row-major bf16.
__device__ __forceinline__ v16bf load_frag(const __bf16* base, int ld, int row0,
                                           int k0, int lane) {
  int hb = (lane >> 4) * 8;
  int mm = lane & 15;
  const __bf16* p = base + (size_t)(row0 + mm) * ld + k0 + hb;
  v16bf f;
#pragma unroll
  for (int e = 0; e < 8; ++e) f[e] = p[e];
#pragma unroll
  for (int e = 0; e < 8; ++e) f[8 + e] = p[16 + e];
  return f;
}

// ---------------------------------------------------------------------------
// T5 relative position bucket — integer-only (no transcendentals).
// large = 8 + trunc(log(a/8)/log(16)*8) = 2 + floor(2*log2(a)) for a >= 8.
// ---------------------------------------------------------------------------
__device__ __forceinline__ int rel_bucket(int rel) {
  int rb = (rel > 0) ? 16 : 0;
  int a  = (rel < 0) ? -rel : rel;
  if (a < 8) return rb + a;
  int e  = 31 - __clz(a);                               // floor(log2 a)
  long long aa = (long long)a * (long long)a;
  int f2 = 2 * e + ((aa >> (2 * e + 1)) ? 1 : 0);       // floor(2*log2 a)
  int lg = 2 + f2;
  if (lg > 15) lg = 15;
  return rb + lg;
}

// ---------------------------------------------------------------------------
// fp32 -> bf16 conversion (vectorized)
// ---------------------------------------------------------------------------
__global__ __launch_bounds__(256) void cvt_kernel(const float* __restrict__ s,
                                                  __bf16* __restrict__ d, int n4) {
  int i = blockIdx.x * 256 + threadIdx.x;
  if (i < n4) {
    v4f x = *(const v4f*)&s[(size_t)i * 4];
    __bf16* p = d + (size_t)i * 4;
    p[0] = (__bf16)x[0]; p[1] = (__bf16)x[1];
    p[2] = (__bf16)x[2]; p[3] = (__bf16)x[3];
  }
}

// src [rows][cols] fp32 -> dst [cols][rows] bf16, 32x32 LDS tile transpose
// (both read and write fully coalesced)
__global__ __launch_bounds__(256) void cvtT_kernel(const float* __restrict__ s,
                                                   __bf16* __restrict__ d,
                                                   int rows, int cols) {
  __shared__ __bf16 tile[32][33];
  int c0 = blockIdx.x * 32;
  int r0 = blockIdx.y * 32;
  int tx = threadIdx.x & 31;
  int ty = threadIdx.x >> 5;          // 8 row-groups
#pragma unroll
  for (int i = 0; i < 4; ++i) {
    int r = ty + i * 8;
    tile[tx][r] = (__bf16)s[(size_t)(r0 + r) * cols + c0 + tx];
  }
  __syncthreads();
#pragma unroll
  for (int i = 0; i < 4; ++i) {
    int c = ty + i * 8;
    d[(size_t)(c0 + c) * rows + r0 + tx] = tile[c][tx];
  }
}

// ---------------------------------------------------------------------------
// position_bias output: [16][2048][2048] fp32; float4 stores, table in LDS
// ---------------------------------------------------------------------------
__global__ __launch_bounds__(256) void bias_kernel(const float* __restrict__ tbl,
                                                   float* __restrict__ out) {
  __shared__ float t[NHEAD * 32];
  t[threadIdx.x]       = tbl[threadIdx.x];
  t[threadIdx.x + 256] = tbl[threadIdx.x + 256];
  __syncthreads();

  int idx4 = blockIdx.x * 256 + threadIdx.x;   // over 2048 * 512
  int q  = idx4 >> 9;
  int k0 = (idx4 & 511) * 4;
  int bk[4];
#pragma unroll
  for (int i = 0; i < 4; ++i) bk[i] = rel_bucket(k0 + i - q) * NHEAD;
  size_t base = (size_t)q * SEQ + k0;
#pragma unroll
  for (int h = 0; h < NHEAD; ++h) {
    v4f v = { t[bk[0] + h], t[bk[1] + h], t[bk[2] + h], t[bk[3] + h] };
    *(v4f*)&out[(size_t)h * (SEQ * SEQ) + base] = v;
  }
}

// ---------------------------------------------------------------------------
// QKV projection (double-buffered LDS, async staging): C[4096][3072] -> Q/K/V
// ---------------------------------------------------------------------------
__global__ __launch_bounds__(256) void gemm_qkv_kernel(
    const __bf16* __restrict__ A, const __bf16* __restrict__ Bt,
    __bf16* __restrict__ Qo, __bf16* __restrict__ Ko, __bf16* __restrict__ Vo) {
  __shared__ __attribute__((aligned(16))) __bf16 As[2][128][32];
  __shared__ __attribute__((aligned(16))) __bf16 Bs[2][128][32];

  int wid  = threadIdx.x >> 5;
  int lane = threadIdx.x & 31;
  int hf   = lane >> 4;
  int n    = lane & 15;
  int waveM = wid >> 2;
  int waveN = wid & 3;
  int rowBase = blockIdx.x * 128;
  int colBase = blockIdx.y * 128;

  int c0 = threadIdx.x * 2, c1 = c0 + 1;
  int lr0 = c0 >> 2, lo0 = (c0 & 3) * 8;
  int lr1 = c1 >> 2, lo1 = (c1 & 3) * 8;

  auto astage = [&](int buf, int k0) {
    async_ld_b128(lds_off(&As[buf][lr0][lo0]),
                  &A[(size_t)(rowBase + lr0) * DMODEL + k0 + lo0]);
    async_ld_b128(lds_off(&As[buf][lr1][lo1]),
                  &A[(size_t)(rowBase + lr1) * DMODEL + k0 + lo1]);
    async_ld_b128(lds_off(&Bs[buf][lr0][lo0]),
                  &Bt[(size_t)(colBase + lr0) * DMODEL + k0 + lo0]);
    async_ld_b128(lds_off(&Bs[buf][lr1][lo1]),
                  &Bt[(size_t)(colBase + lr1) * DMODEL + k0 + lo1]);
  };

  v8f acc[4][2];
#pragma unroll
  for (int i = 0; i < 4; ++i)
#pragma unroll
    for (int j = 0; j < 2; ++j) acc[i][j] = (v8f)0.0f;

  astage(0, 0);
  wait_async0();
  __syncthreads();

  for (int step = 0; step < DMODEL / 32; ++step) {
    int cur = step & 1;
    bool more = (step + 1) < DMODEL / 32;
    if (more) astage(cur ^ 1, (step + 1) * 32);   // DMA next tile during WMMAs

    v16bf af[4], bfr[2];
#pragma unroll
    for (int i = 0; i < 4; ++i)
      af[i] = load_frag(&As[cur][0][0], 32, waveM * 64 + i * 16, 0, lane);
#pragma unroll
    for (int j = 0; j < 2; ++j)
      bfr[j] = load_frag(&Bs[cur][0][0], 32, waveN * 32 + j * 16, 0, lane);
#pragma unroll
    for (int i = 0; i < 4; ++i)
#pragma unroll
      for (int j = 0; j < 2; ++j) acc[i][j] = wmma_bf16(af[i], bfr[j], acc[i][j]);

    wait_async0();
    __syncthreads();
  }

#pragma unroll
  for (int i = 0; i < 4; ++i)
#pragma unroll
    for (int j = 0; j < 2; ++j)
#pragma unroll
      for (int r = 0; r < 8; ++r) {
        int m   = r + hf * 8;
        int row = rowBase + waveM * 64 + i * 16 + m;
        int col = colBase + waveN * 32 + j * 16 + n;
        int which = col >> 10;
        int rem   = col & 1023;
        int hh = rem >> 6;
        int dd = rem & 63;
        int bb = row >> 11;
        int ss = row & 2047;
        __bf16* dst = (which == 0) ? Qo : (which == 1) ? Ko : Vo;
        dst[(((size_t)bb * NHEAD + hh) * SEQ + ss) * DHEAD + dd] = (__bf16)acc[i][j][r];
      }
}

// ---------------------------------------------------------------------------
// Flash attention: WG = (b,h, 128 q-rows); 8 waves x 16 q-rows.
// K chunks of 64, double-buffered LDS; K staged via async DMA, V via
// register transpose.
// ---------------------------------------------------------------------------
__global__ __launch_bounds__(256) void attn_kernel(
    const __bf16* __restrict__ Qg, const __bf16* __restrict__ Kg,
    const __bf16* __restrict__ Vg, const float* __restrict__ tbl,
    __bf16* __restrict__ Og) {
  __shared__ __attribute__((aligned(16))) __bf16 Kc[2][64][64];
  __shared__ __attribute__((aligned(16))) __bf16 Vt[2][64][64];
  __shared__ __attribute__((aligned(16))) __bf16 Pld[8][16][64];
  __shared__ float biasv[32];

  int bh   = blockIdx.y;
  int h    = bh & 15;
  int b    = bh >> 4;
  int wid  = threadIdx.x >> 5;
  int lane = threadIdx.x & 31;
  int hf   = lane >> 4;
  int n    = lane & 15;
  int q0   = blockIdx.x * 128 + wid * 16;

  const __bf16* Qh = Qg + (size_t)bh * SEQ * DHEAD;
  const __bf16* Kh = Kg + (size_t)bh * SEQ * DHEAD;
  const __bf16* Vh = Vg + (size_t)bh * SEQ * DHEAD;

  if (threadIdx.x < 32) biasv[threadIdx.x] = tbl[threadIdx.x * NHEAD + h];

  int c0 = threadIdx.x * 2, c1 = c0 + 1;
  int sr0 = c0 >> 3, so0 = (c0 & 7) * 8;
  int sr1 = c1 >> 3, so1 = (c1 & 7) * 8;

  auto k_async = [&](int buf, int kc) {
    async_ld_b128(lds_off(&Kc[buf][sr0][so0]),
                  &Kh[(size_t)(kc + sr0) * DHEAD + so0]);
    async_ld_b128(lds_off(&Kc[buf][sr1][so1]),
                  &Kh[(size_t)(kc + sr1) * DHEAD + so1]);
  };
  auto v_load = [&](int kc, v4u& va, v4u& vb) {
    va = *(const v4u*)&Vh[(size_t)(kc + sr0) * DHEAD + so0];
    vb = *(const v4u*)&Vh[(size_t)(kc + sr1) * DHEAD + so1];
  };
  auto v_store = [&](int buf, v4u va, v4u vb) {
    RB u0, u1; u0.u = va; u1.u = vb;
#pragma unroll
    for (int j = 0; j < 8; ++j) {
      Vt[buf][so0 + j][sr0] = u0.e[j];   // Vt[d][k]
      Vt[buf][so1 + j][sr1] = u1.e[j];
    }
  };

  v16bf qf0 = load_frag(Qh, DHEAD, q0, 0, lane);
  v16bf qf1 = load_frag(Qh, DHEAD, q0, 32, lane);

  v8f o[4];
#pragma unroll
  for (int j = 0; j < 4; ++j) o[j] = (v8f)0.0f;
  float mrow[8], lrow[8];
#pragma unroll
  for (int r = 0; r < 8; ++r) { mrow[r] = -3.0e38f; lrow[r] = 0.0f; }

  v4u rva, rvb;
  k_async(0, 0);
  v_load(0, rva, rvb);
  v_store(0, rva, rvb);
  wait_async0();
  __syncthreads();

  for (int step = 0; step < SEQ / 64; ++step) {
    int cur = step & 1;
    int kc  = step * 64;
    bool more = (step + 1) < SEQ / 64;
    if (more) {
      k_async(cur ^ 1, kc + 64);        // DMA next K tile during compute
      v_load(kc + 64, rva, rvb);
    }

    // S = Q K^T : 16 x 64 (4 accumulators)
    v8f s[4];
#pragma unroll
    for (int t = 0; t < 4; ++t) {
      s[t] = (v8f)0.0f;
      v16bf kf = load_frag(&Kc[cur][0][0], 64, t * 16, 0, lane);
      s[t] = wmma_bf16(qf0, kf, s[t]);
      kf = load_frag(&Kc[cur][0][0], 64, t * 16, 32, lane);
      s[t] = wmma_bf16(qf1, kf, s[t]);
    }

    // bias: wave-uniform saturation fast path (bucket saturates at |rel|>=91)
    bool uniPos = (kc - (q0 + 15)) >= 91;
    bool uniNeg = (q0 - (kc + 63)) >= 91;
    if (uniPos || uniNeg) {
      float bunif = uniPos ? biasv[31] : biasv[15];
#pragma unroll
      for (int t = 0; t < 4; ++t)
#pragma unroll
        for (int r = 0; r < 8; ++r) s[t][r] += bunif;
    } else {
#pragma unroll
      for (int r = 0; r < 8; ++r) {
        int q = q0 + r + hf * 8;
#pragma unroll
        for (int t = 0; t < 4; ++t)
          s[t][r] += biasv[rel_bucket(kc + t * 16 + n - q)];
      }
    }

    // online softmax over the 64-wide chunk
#pragma unroll
    for (int r = 0; r < 8; ++r) {
      float mx = fmaxf(fmaxf(s[0][r], s[1][r]), fmaxf(s[2][r], s[3][r]));
      mx = fmaxf(mx, __shfl_xor(mx, 1, 16));
      mx = fmaxf(mx, __shfl_xor(mx, 2, 16));
      mx = fmaxf(mx, __shfl_xor(mx, 4, 16));
      mx = fmaxf(mx, __shfl_xor(mx, 8, 16));
      float mnew = fmaxf(mrow[r], mx);
      float sc   = __expf(mrow[r] - mnew);
      mrow[r] = mnew;
      float p0 = __expf(s[0][r] - mnew);
      float p1 = __expf(s[1][r] - mnew);
      float p2 = __expf(s[2][r] - mnew);
      float p3 = __expf(s[3][r] - mnew);
      float rs = (p0 + p1) + (p2 + p3);
      rs += __shfl_xor(rs, 1, 16);
      rs += __shfl_xor(rs, 2, 16);
      rs += __shfl_xor(rs, 4, 16);
      rs += __shfl_xor(rs, 8, 16);
      lrow[r] = lrow[r] * sc + rs;
#pragma unroll
      for (int j = 0; j < 4; ++j) o[j][r] *= sc;
      int m = r + hf * 8;
      Pld[wid][m][n]      = (__bf16)p0;
      Pld[wid][m][16 + n] = (__bf16)p1;
      Pld[wid][m][32 + n] = (__bf16)p2;
      Pld[wid][m][48 + n] = (__bf16)p3;
    }

    // O += P V (wave-private P through LDS: C-layout -> A-layout)
    v16bf pf0 = load_frag(&Pld[wid][0][0], 64, 0, 0, lane);
    v16bf pf1 = load_frag(&Pld[wid][0][0], 64, 0, 32, lane);
#pragma unroll
    for (int j = 0; j < 4; ++j) {
      v16bf vf = load_frag(&Vt[cur][0][0], 64, j * 16, 0, lane);
      o[j] = wmma_bf16(pf0, vf, o[j]);
      vf = load_frag(&Vt[cur][0][0], 64, j * 16, 32, lane);
      o[j] = wmma_bf16(pf1, vf, o[j]);
    }

    if (more) v_store(cur ^ 1, rva, rvb);
    wait_async0();
    __syncthreads();
  }

  // normalize + store O bf16 row-major [4096][1024], col = h*64 + d
#pragma unroll
  for (int r = 0; r < 8; ++r) {
    float inv = 1.0f / lrow[r];
    int m = r + hf * 8;
    size_t row = (size_t)b * SEQ + q0 + m;
    __bf16* dst = Og + row * DMODEL + h * DHEAD;
#pragma unroll
    for (int j = 0; j < 4; ++j) dst[j * 16 + n] = (__bf16)(o[j][r] * inv);
  }
}

// ---------------------------------------------------------------------------
// Output projection (double-buffered, async staging): out[4096][1024] fp32
// ---------------------------------------------------------------------------
__global__ __launch_bounds__(256) void gemm_out_kernel(
    const __bf16* __restrict__ A, const __bf16* __restrict__ Bt,
    float* __restrict__ Co) {
  __shared__ __attribute__((aligned(16))) __bf16 As[2][128][32];
  __shared__ __attribute__((aligned(16))) __bf16 Bs[2][128][32];

  int wid  = threadIdx.x >> 5;
  int lane = threadIdx.x & 31;
  int hf   = lane >> 4;
  int n    = lane & 15;
  int waveM = wid >> 2;
  int waveN = wid & 3;
  int rowBase = blockIdx.x * 128;
  int colBase = blockIdx.y * 128;

  int c0 = threadIdx.x * 2, c1 = c0 + 1;
  int lr0 = c0 >> 2, lo0 = (c0 & 3) * 8;
  int lr1 = c1 >> 2, lo1 = (c1 & 3) * 8;

  auto astage = [&](int buf, int k0) {
    async_ld_b128(lds_off(&As[buf][lr0][lo0]),
                  &A[(size_t)(rowBase + lr0) * DMODEL + k0 + lo0]);
    async_ld_b128(lds_off(&As[buf][lr1][lo1]),
                  &A[(size_t)(rowBase + lr1) * DMODEL + k0 + lo1]);
    async_ld_b128(lds_off(&Bs[buf][lr0][lo0]),
                  &Bt[(size_t)(colBase + lr0) * DMODEL + k0 + lo0]);
    async_ld_b128(lds_off(&Bs[buf][lr1][lo1]),
                  &Bt[(size_t)(colBase + lr1) * DMODEL + k0 + lo1]);
  };

  v8f acc[4][2];
#pragma unroll
  for (int i = 0; i < 4; ++i)
#pragma unroll
    for (int j = 0; j < 2; ++j) acc[i][j] = (v8f)0.0f;

  astage(0, 0);
  wait_async0();
  __syncthreads();

  for (int step = 0; step < DMODEL / 32; ++step) {
    int cur = step & 1;
    bool more = (step + 1) < DMODEL / 32;
    if (more) astage(cur ^ 1, (step + 1) * 32);

    v16bf af[4], bfr[2];
#pragma unroll
    for (int i = 0; i < 4; ++i)
      af[i] = load_frag(&As[cur][0][0], 32, waveM * 64 + i * 16, 0, lane);
#pragma unroll
    for (int j = 0; j < 2; ++j)
      bfr[j] = load_frag(&Bs[cur][0][0], 32, waveN * 32 + j * 16, 0, lane);
#pragma unroll
    for (int i = 0; i < 4; ++i)
#pragma unroll
      for (int j = 0; j < 2; ++j) acc[i][j] = wmma_bf16(af[i], bfr[j], acc[i][j]);

    wait_async0();
    __syncthreads();
  }

#pragma unroll
  for (int i = 0; i < 4; ++i)
#pragma unroll
    for (int j = 0; j < 2; ++j)
#pragma unroll
      for (int r = 0; r < 8; ++r) {
        int m   = r + hf * 8;
        int row = rowBase + waveM * 64 + i * 16 + m;
        int col = colBase + waveN * 32 + j * 16 + n;
        Co[(size_t)row * DMODEL + col] = acc[i][j][r];
      }
}

// ---------------------------------------------------------------------------
// Host entry
// ---------------------------------------------------------------------------
extern "C" void kernel_launch(void* const* d_in, const int* in_sizes, int n_in,
                              void* d_out, int out_size, void* d_ws, size_t ws_size,
                              hipStream_t stream) {
  (void)in_sizes; (void)n_in; (void)out_size; (void)ws_size;
  const float* hs   = (const float*)d_in[0];
  const float* wqkv = (const float*)d_in[1];
  const float* wo   = (const float*)d_in[2];
  const float* tbl  = (const float*)d_in[3];
  float* out = (float*)d_out;

  __bf16* hsb   = (__bf16*)d_ws;
  __bf16* wqkvT = hsb   + (size_t)NROWS * DMODEL;
  __bf16* woT   = wqkvT + (size_t)3 * DMODEL * DMODEL;
  __bf16* Qb    = woT   + (size_t)DMODEL * DMODEL;
  __bf16* Kb    = Qb    + (size_t)2 * NHEAD * SEQ * DHEAD;
  __bf16* Vb    = Kb    + (size_t)2 * NHEAD * SEQ * DHEAD;
  __bf16* Ob    = Vb    + (size_t)2 * NHEAD * SEQ * DHEAD;

  cvt_kernel <<<dim3(4096), dim3(256), 0, stream>>>(hs, hsb, (NROWS * DMODEL) / 4);
  cvtT_kernel<<<dim3(96, 32), dim3(256), 0, stream>>>(wqkv, wqkvT, DMODEL, 3 * DMODEL);
  cvtT_kernel<<<dim3(32, 32), dim3(256), 0, stream>>>(wo, woT, DMODEL, DMODEL);

  bias_kernel<<<dim3((SEQ * (SEQ / 4)) / 256), dim3(256), 0, stream>>>(
      tbl, out + (size_t)NROWS * DMODEL);

  gemm_qkv_kernel<<<dim3(32, 24), dim3(256), 0, stream>>>(hsb, wqkvT, Qb, Kb, Vb);
  attn_kernel    <<<dim3(16, 32), dim3(256), 0, stream>>>(Qb, Kb, Vb, tbl, Ob);
  gemm_out_kernel<<<dim3(32, 8),  dim3(256), 0, stream>>>(Ob, woT, out);
}
